// MultiHeadAttention_15144054685789
// MI455X (gfx1250) — matrix operand order repaired and checked by reading, not compiled, not run on hardware
//
#include <hip/hip_runtime.h>

// Problem constants (reference: B=2, S=2048, D=1024, H=16, HD=64)
#define B_DIM 2
#define S_LEN 2048
#define D_DIM 1024
#define H_NUM 16
#define HD_DIM 64
#define M_ROWS (B_DIM * S_LEN) // 4096

typedef __bf16 bhalf;
typedef __attribute__((ext_vector_type(16))) __bf16 v16bf;
typedef __attribute__((ext_vector_type(8)))  float  v8f;
typedef __attribute__((ext_vector_type(2)))  float  vf2;
typedef __attribute__((ext_vector_type(2)))  __bf16 vbf2;

union FragBF { uint4 u4[2]; v16bf v; };

#define WMMA_BF16(A_, B_, C_) \
    __builtin_amdgcn_wmma_f32_16x16x32_bf16(false, (A_), false, (B_), (short)0, (C_), false, false)

// Native bf16 converts (gfx1250: v_cvt_pk_bf16_f32).
__device__ __forceinline__ bhalf f2bf(float x) { return (bhalf)x; }
__device__ __forceinline__ unsigned pack2bf(float a, float b) {
    vf2 t; t.x = a; t.y = b;   // one fptrunc <2 x float> -> v_cvt_pk_bf16_f32
    return __builtin_bit_cast(unsigned, __builtin_convertvector(t, vbf2));
}

// CDNA5 async global->LDS copy (16B/lane), tracked by ASYNCcnt.
// VDST = LDS byte address (low 32 bits of generic pointer to __shared__),
// VADDR = 64-bit global address.
__device__ __forceinline__ void async_copy_b128(unsigned lds_off, const void* gptr) {
    asm volatile("global_load_async_to_lds_b128 %0, %1, off"
                 :: "v"(lds_off), "v"(gptr) : "memory");
}
__device__ __forceinline__ void async_wait0() {
    asm volatile("s_wait_asynccnt 0x0" ::: "memory");
}

// ---------------------------------------------------------------------------
// GEMM tile config: 128x64 block tile, k-step 32 (WMMA bf16 K), 8 waves/block
// ---------------------------------------------------------------------------
#define BM 128
#define BN 64
#define BK 32
#define LDA (BK + 8)  // 40 bf16 = 80B rows (16B aligned)
#define LDB (BK + 8)

// C = X(f32)[M,K] * W(f32)[K,N] -> bf16 [B,H,S,HD], scaled. Double-buffered LDS.
__global__ __launch_bounds__(256)
void qkv_gemm_kernel(const float* __restrict__ X, const float* __restrict__ W,
                     bhalf* __restrict__ Out, float scale)
{
    __shared__ __align__(16) bhalf As[2][BM][LDA];
    __shared__ __align__(16) bhalf BsT[2][BN][LDB]; // W tile transposed: [n][k]

    const int tid  = threadIdx.x;
    const int lane = tid & 31;
    const int wid  = tid >> 5;
    const int m0   = blockIdx.x * BM;
    const int n0   = blockIdx.y * BN;   // == head * HD (BN == HD)

    v8f acc[4];
    for (int c = 0; c < 4; ++c) acc[c] = {};

    const int arow = tid >> 1;          // A: 2 threads/row, 16 f32 each
    const int acol = (tid & 1) * 16;
    const int kp   = tid & 15;          // W: k-pair index (k = 2*kp, 2*kp+1)
    const int ngrp = tid >> 4;          // W: n-group (n = ngrp*4 .. +3)

    const float* Xrow = X + (size_t)(m0 + arow) * D_DIM;
    const float* Wrow = W + (size_t)(2 * kp) * D_DIM + n0 + ngrp * 4;

    const int rlo   = lane & 15;
    const int khalf = lane >> 4;
    const int wm    = wid * 16;

    float4 xr[4], w0r, w1r;
    auto load_tile = [&](int k0) {
        const float4* xv = reinterpret_cast<const float4*>(Xrow + k0 + acol);
#pragma unroll
        for (int j = 0; j < 4; ++j) xr[j] = xv[j];
        const float* wp = Wrow + (size_t)k0 * D_DIM;
        w0r = *reinterpret_cast<const float4*>(wp);
        w1r = *reinterpret_cast<const float4*>(wp + D_DIM);
    };
    auto store_tile = [&](int buf) {
#pragma unroll
        for (int j = 0; j < 4; ++j)
            *reinterpret_cast<uint2*>(&As[buf][arow][acol + 4 * j]) =
                make_uint2(pack2bf(xr[j].x, xr[j].y), pack2bf(xr[j].z, xr[j].w));
        const float a0[4] = {w0r.x, w0r.y, w0r.z, w0r.w};
        const float a1[4] = {w1r.x, w1r.y, w1r.z, w1r.w};
#pragma unroll
        for (int j = 0; j < 4; ++j) // both k's of the pair share one b32 word
            *reinterpret_cast<unsigned*>(&BsT[buf][ngrp * 4 + j][2 * kp]) =
                pack2bf(a0[j], a1[j]);
    };

    load_tile(0);
    store_tile(0);
    __syncthreads();

    for (int k0 = 0; k0 < D_DIM; k0 += BK) {
        const int p = (k0 >> 5) & 1;
        const bool nxt = (k0 + BK) < D_DIM;
        if (nxt) load_tile(k0 + BK);
        if (k0 + 2 * BK < D_DIM) { // distance-2 prefetch -> global_prefetch_b8
            __builtin_prefetch(Xrow + k0 + 2 * BK + acol, 0, 1);
            __builtin_prefetch(Wrow + (size_t)(k0 + 2 * BK) * D_DIM, 0, 1);
        }

        FragBF a;
        a.u4[0] = *reinterpret_cast<const uint4*>(&As[p][wm + rlo][khalf * 16]);
        a.u4[1] = *reinterpret_cast<const uint4*>(&As[p][wm + rlo][khalf * 16 + 8]);
#pragma unroll
        for (int c = 0; c < 4; ++c) {
            FragBF b;
            const bhalf* bp = &BsT[p][c * 16 + rlo][khalf * 16];
            b.u4[0] = reinterpret_cast<const uint4*>(bp)[0];
            b.u4[1] = reinterpret_cast<const uint4*>(bp)[1];
            acc[c] = WMMA_BF16(a.v, b.v, acc[c]);
        }
        if (nxt) store_tile(p ^ 1);
        __syncthreads();
    }

    // Epilogue: C-frag layout -> [B,H,S,HD] bf16
    const unsigned h = (unsigned)n0 / HD_DIM;
    const int rowoff = khalf * 8;
#pragma unroll
    for (int c = 0; c < 4; ++c) {
#pragma unroll
        for (int rr = 0; rr < 8; ++rr) {
            unsigned m  = (unsigned)(m0 + wm + rowoff + rr);
            unsigned bb = m >> 11, s = m & (S_LEN - 1);
            unsigned hd = c * 16 + rlo;
            Out[(((size_t)bb * H_NUM + h) * S_LEN + s) * HD_DIM + hd] = f2bf(acc[c][rr] * scale);
        }
    }
}

// out = A(bf16)[M,K] * W(f32)[K,N] + bias -> fp32 [M,N].
// A tile staged with async global->LDS; double-buffered.
__global__ __launch_bounds__(256)
void out_gemm_kernel(const bhalf* __restrict__ A, const float* __restrict__ W,
                     const float* __restrict__ bias, float* __restrict__ Out)
{
    __shared__ __align__(16) bhalf As[2][BM][LDA];
    __shared__ __align__(16) bhalf BsT[2][BN][LDB];

    const int tid  = threadIdx.x;
    const int lane = tid & 31;
    const int wid  = tid >> 5;
    const int m0   = blockIdx.x * BM;
    const int n0   = blockIdx.y * BN;

    v8f acc[4];
    for (int c = 0; c < 4; ++c) acc[c] = {};

    const int arow = tid >> 1;
    const int acol = (tid & 1) * 16;
    const int kp   = tid & 15;
    const int ngrp = tid >> 4;

    const bhalf* Arow = A + (size_t)(m0 + arow) * D_DIM;
    const float* Wrow = W + (size_t)(2 * kp) * D_DIM + n0 + ngrp * 4;

    const int rlo   = lane & 15;
    const int khalf = lane >> 4;
    const int wm    = wid * 16;

    float4 w0r, w1r;
    auto stage_a_async = [&](int k0, int buf) {
        unsigned l0 = (unsigned)(size_t)&As[buf][arow][acol];
        async_copy_b128(l0,      Arow + k0 + acol);
        async_copy_b128(l0 + 16, Arow + k0 + acol + 8);
    };
    auto load_w = [&](int k0) {
        const float* wp = Wrow + (size_t)k0 * D_DIM;
        w0r = *reinterpret_cast<const float4*>(wp);
        w1r = *reinterpret_cast<const float4*>(wp + D_DIM);
    };
    auto store_w = [&](int buf) {
        const float a0[4] = {w0r.x, w0r.y, w0r.z, w0r.w};
        const float a1[4] = {w1r.x, w1r.y, w1r.z, w1r.w};
#pragma unroll
        for (int j = 0; j < 4; ++j)
            *reinterpret_cast<unsigned*>(&BsT[buf][ngrp * 4 + j][2 * kp]) =
                pack2bf(a0[j], a1[j]);
    };

    stage_a_async(0, 0);
    load_w(0);
    store_w(0);
    async_wait0();
    __syncthreads();

    for (int k0 = 0; k0 < D_DIM; k0 += BK) {
        const int p = (k0 >> 5) & 1;
        const bool nxt = (k0 + BK) < D_DIM;
        if (nxt) {
            stage_a_async(k0 + BK, p ^ 1);
            load_w(k0 + BK);
        }
        if (k0 + 2 * BK < D_DIM)
            __builtin_prefetch(Wrow + (size_t)(k0 + 2 * BK) * D_DIM, 0, 1);

        FragBF a;
        a.u4[0] = *reinterpret_cast<const uint4*>(&As[p][wm + rlo][khalf * 16]);
        a.u4[1] = *reinterpret_cast<const uint4*>(&As[p][wm + rlo][khalf * 16 + 8]);
#pragma unroll
        for (int c = 0; c < 4; ++c) {
            FragBF b;
            const bhalf* bp = &BsT[p][c * 16 + rlo][khalf * 16];
            b.u4[0] = reinterpret_cast<const uint4*>(bp)[0];
            b.u4[1] = reinterpret_cast<const uint4*>(bp)[1];
            acc[c] = WMMA_BF16(a.v, b.v, acc[c]);
        }
        if (nxt) store_w(p ^ 1);
        async_wait0();
        __syncthreads();
    }

    const int rowoff = khalf * 8;
#pragma unroll
    for (int c = 0; c < 4; ++c) {
        float bv = bias[n0 + c * 16 + rlo];
#pragma unroll
        for (int rr = 0; rr < 8; ++rr) {
            unsigned m = (unsigned)(m0 + wm + rowoff + rr);
            Out[(size_t)m * D_DIM + n0 + c * 16 + rlo] = acc[c][rr] + bv;
        }
    }
}

// ---------------------------------------------------------------------------
// Flash attention: block = (b,h, 128-query chunk); 8 waves x 16 query rows.
// K tile: async global->LDS (double-buffered). V tile: global->reg, scattered
// transposed into LDS after the WMMA block. Q pre-scaled by 1/sqrt(HD). Causal.
// ---------------------------------------------------------------------------
#define QT 128
#define KT 128
#define LDK (HD_DIM + 8)  // 72 bf16 = 144B rows
#define LDV (KT + 8)      // 136 bf16 = 272B rows
#define LDP (32 + 8)      // 40 bf16 = 80B rows

__global__ __launch_bounds__(256)
void attn_kernel(const bhalf* __restrict__ Q, const bhalf* __restrict__ K,
                 const bhalf* __restrict__ V, bhalf* __restrict__ O)
{
    __shared__ __align__(16) bhalf Ks[2][KT][LDK];       // [key][d]
    __shared__ __align__(16) bhalf VsT[2][HD_DIM][LDV];  // [d][key]
    __shared__ __align__(16) bhalf Ps[8][16][LDP];       // per-wave P staging

    const int tid  = threadIdx.x;
    const int lane = tid & 31;
    const int wid  = tid >> 5;
    const int bh   = blockIdx.y;                 // b*H + h
    const int h    = bh % H_NUM;
    const int bb   = bh / H_NUM;
    const int q0   = blockIdx.x * QT;
    const size_t base = (size_t)bh * S_LEN * HD_DIM;

    const int qw    = q0 + wid * 16;
    const int rlo   = lane & 15;
    const int khalf = lane >> 4;

    // Q fragments (d 0..31 and 32..63) straight from global bf16
    FragBF aq0, aq1;
    {
        const bhalf* qr = Q + base + (size_t)(qw + rlo) * HD_DIM;
        aq0.u4[0] = reinterpret_cast<const uint4*>(qr + khalf * 16)[0];
        aq0.u4[1] = reinterpret_cast<const uint4*>(qr + khalf * 16 + 8)[0];
        aq1.u4[0] = reinterpret_cast<const uint4*>(qr + 32 + khalf * 16)[0];
        aq1.u4[1] = reinterpret_cast<const uint4*>(qr + 32 + khalf * 16 + 8)[0];
    }

    v8f o[4];
    for (int c = 0; c < 4; ++c) o[c] = {};
    float m_r[8], l_r[8];
#pragma unroll
    for (int r = 0; r < 8; ++r) { m_r[r] = -3.0e38f; l_r[r] = 0.0f; }

    uint4 vr[4];
    auto stage_k = [&](int kb, int buf) { // 4 async b128 per thread (16KB tile)
#pragma unroll
        for (int j = 0; j < 4; ++j) {
            int idx = tid + j * 256;           // 1024 x 16B
            int row = idx >> 3, col = (idx & 7) * 8;
            unsigned loff = (unsigned)(size_t)&Ks[buf][row][col];
            async_copy_b128(loff, K + base + (size_t)(kb + row) * HD_DIM + col);
        }
    };
    auto load_v = [&](int kb) {
        const uint4* vsrc = reinterpret_cast<const uint4*>(V + base + (size_t)kb * HD_DIM);
#pragma unroll
        for (int j = 0; j < 4; ++j) vr[j] = vsrc[tid + j * 256];
    };
    auto scatter_v = [&](int buf) {
#pragma unroll
        for (int j = 0; j < 4; ++j) {
            int idx = tid + j * 256;
            int row = idx >> 3, col = (idx & 7) * 8;
            union { uint4 q; unsigned short s[8]; } u; u.q = vr[j];
#pragma unroll
            for (int e = 0; e < 8; ++e)
                VsT[buf][col + e][row] = __builtin_bit_cast(bhalf, u.s[e]);
        }
    };

    const int nkt = blockIdx.x + 1; // causal: key tiles 0..q-tile
    // prologue: stage tile 0 into buffer 0
    stage_k(0, 0);
    load_v(0);
    scatter_v(0);
    async_wait0();
    __syncthreads();

    for (int kt = 0; kt < nkt; ++kt) {
        const int p = kt & 1;
        const bool nxt = (kt + 1) < nkt;
        const int kbase0 = kt * KT;
        if (nxt) {
            stage_k(kbase0 + KT, p ^ 1); // overlaps with WMMA below
            load_v(kbase0 + KT);
        }

        for (int j0 = 0; j0 < KT; j0 += 32) {
            const int kk = kbase0 + j0;
            if (kk > qw + 15) break;          // wave-uniform causal cutoff

            // scores: 16 queries x 32 keys = two C-frags, 2 WMMAs each (d split)
            v8f s0 = {}, s1 = {};
#pragma unroll
            for (int cc = 0; cc < 2; ++cc) {
                const bhalf* kpt = &Ks[p][j0 + cc * 16 + rlo][khalf * 16];
                FragBF bk0, bk1;
                bk0.u4[0] = reinterpret_cast<const uint4*>(kpt)[0];
                bk0.u4[1] = reinterpret_cast<const uint4*>(kpt)[1];
                bk1.u4[0] = reinterpret_cast<const uint4*>(kpt + 32)[0];
                bk1.u4[1] = reinterpret_cast<const uint4*>(kpt + 32)[1];
                v8f sc = {};
                sc = WMMA_BF16(aq0.v, bk0.v, sc);
                sc = WMMA_BF16(aq1.v, bk1.v, sc);
                if (cc == 0) s0 = sc; else s1 = sc;
            }

            // causal mask + online softmax (row stats across 16-lane groups)
            float p0s[8], p1s[8], alpha[8];
#pragma unroll
            for (int r = 0; r < 8; ++r) {
                const int qrow = qw + khalf * 8 + r;
                const int key0 = kk + rlo, key1 = key0 + 16;
                float v0 = (key0 <= qrow) ? s0[r] : -3.0e38f;
                float v1 = (key1 <= qrow) ? s1[r] : -3.0e38f;
                float mt = fmaxf(v0, v1);
#pragma unroll
                for (int off = 1; off < 16; off <<= 1)
                    mt = fmaxf(mt, __shfl_xor(mt, off, 32));
                float mn = fmaxf(m_r[r], mt);
                alpha[r] = __expf(m_r[r] - mn);
                m_r[r] = mn;
                float pe0 = __expf(v0 - mn);
                float pe1 = __expf(v1 - mn);
                float rs = pe0 + pe1;
#pragma unroll
                for (int off = 1; off < 16; off <<= 1)
                    rs += __shfl_xor(rs, off, 32);
                l_r[r] = l_r[r] * alpha[r] + rs;
                p0s[r] = pe0; p1s[r] = pe1;
            }
#pragma unroll
            for (int c = 0; c < 4; ++c)
#pragma unroll
                for (int r = 0; r < 8; ++r) o[c][r] *= alpha[r];

            // P: C-layout -> per-wave LDS -> A-layout fragment
#pragma unroll
            for (int r = 0; r < 8; ++r) {
                Ps[wid][khalf * 8 + r][rlo]      = f2bf(p0s[r]);
                Ps[wid][khalf * 8 + r][16 + rlo] = f2bf(p1s[r]);
            }
            FragBF pa;
            pa.u4[0] = reinterpret_cast<const uint4*>(&Ps[wid][rlo][khalf * 16])[0];
            pa.u4[1] = reinterpret_cast<const uint4*>(&Ps[wid][rlo][khalf * 16 + 8])[0];

            // o += P @ V  (4 hd-tiles)
#pragma unroll
            for (int c = 0; c < 4; ++c) {
                const bhalf* vp = &VsT[p][c * 16 + rlo][j0 + khalf * 16];
                FragBF bv;
                bv.u4[0] = reinterpret_cast<const uint4*>(vp)[0];
                bv.u4[1] = reinterpret_cast<const uint4*>(vp)[1];
                o[c] = WMMA_BF16(pa.v, bv.v, o[c]);
            }
        }

        if (nxt) scatter_v(p ^ 1);
        async_wait0();
        __syncthreads();
    }

    // epilogue: normalize and write bf16 [B*S, D]
#pragma unroll
    for (int r = 0; r < 8; ++r) {
        float inv = 1.0f / l_r[r];
        int qrow = qw + khalf * 8 + r;
        size_t m = (size_t)bb * S_LEN + qrow;
#pragma unroll
        for (int c = 0; c < 4; ++c) {
            int col = h * HD_DIM + c * 16 + rlo;
            O[m * D_DIM + col] = f2bf(o[c][r] * inv);
        }
    }
}

// ---------------------------------------------------------------------------
extern "C" void kernel_launch(void* const* d_in, const int* in_sizes, int n_in,
                              void* d_out, int out_size, void* d_ws, size_t ws_size,
                              hipStream_t stream) {
    const float* x  = (const float*)d_in[0];
    const float* Wq = (const float*)d_in[1];
    const float* Wk = (const float*)d_in[2];
    const float* Wv = (const float*)d_in[3];
    const float* Wo = (const float*)d_in[4];
    const float* bo = (const float*)d_in[5];
    float* out = (float*)d_out;

    // workspace: Q,K,V bf16 [B,H,S,HD] + attn-out bf16 [B*S, D] = 32 MB
    bhalf* qws = (bhalf*)d_ws;
    bhalf* kws = qws + (size_t)M_ROWS * D_DIM;
    bhalf* vws = kws + (size_t)M_ROWS * D_DIM;
    bhalf* ows = vws + (size_t)M_ROWS * D_DIM;

    dim3 gg(M_ROWS / BM, D_DIM / BN); // 32 x 16
    qkv_gemm_kernel<<<gg, 256, 0, stream>>>(x, Wq, qws, 0.125f); // fold 1/sqrt(64)
    qkv_gemm_kernel<<<gg, 256, 0, stream>>>(x, Wk, kws, 1.0f);
    qkv_gemm_kernel<<<gg, 256, 0, stream>>>(x, Wv, vws, 1.0f);

    attn_kernel<<<dim3(S_LEN / QT, B_DIM * H_NUM), 256, 0, stream>>>(qws, kws, vws, ows);

    out_gemm_kernel<<<gg, 256, 0, stream>>>(ows, Wo, bo, out);
}
